// Attention1_19774029431808
// MI455X (gfx1250) — matrix operand order repaired and checked by reading, not compiled, not run on hardware
//
#include <hip/hip_runtime.h>
#include <hip/hip_bf16.h>
#include <stdint.h>

// ---------------------------------------------------------------------------
// CDNA5 (gfx1250) bf16 WMMA attention pipeline.
//  - all matmuls: v_wmma_f32_16x16x32_bf16 (wave32)
//  - GEMMs: 32x64 output per wave (2x4 register blocking)
//  - attention: TDM (tensor_load_to_lds) stages K/V tiles in LDS, shared by
//    all 4 waves of the workgroup; s_wait_tensorcnt + barrier publishes them.
// ---------------------------------------------------------------------------

typedef __attribute__((ext_vector_type(16))) __bf16 v16bf;
typedef __attribute__((ext_vector_type(8)))  float  v8f;
typedef __attribute__((ext_vector_type(4)))  unsigned int u32x4;
typedef __attribute__((ext_vector_type(8)))  int    i32x8;
typedef __attribute__((ext_vector_type(4)))  int    i32x4;

union FragAB {                 // 32 bytes: one WMMA A or B operand (bf16)
    v16bf v;
    uint4 q[2];
};
union FragC {                  // 32 bytes: one WMMA C/D operand (f32)
    v8f   v;
    float f[8];
};

__device__ __forceinline__ unsigned short f32_to_bf16(float x) {
    union { float f; unsigned u; } a; a.f = x;
    unsigned r = a.u + 0x7FFFu + ((a.u >> 16) & 1u);   // round-to-nearest-even
    return (unsigned short)(r >> 16);
}

// Issue a 2D TDM tile load (bf16 elements) into LDS.  Dims/strides in elements.
__device__ __forceinline__ void tdm_load_2d(unsigned lds_off, const void* gptr,
                                            unsigned tensor_d0, unsigned tensor_d1,
                                            unsigned tile_d0, unsigned tile_d1,
                                            unsigned stride0) {
    unsigned long long ga = (unsigned long long)(size_t)gptr;
    u32x4 g0 = { 1u,                                   // count=1, user desc
                 lds_off,                              // lds_addr (bytes)
                 (unsigned)ga,                         // global_addr[31:0]
                 (unsigned)((ga >> 32) & 0x1FFFFFFu) | (2u << 30) };  // [56:32] | type=2
    i32x8 g1 = { (int)(1u << 16),                                    // data_size=1 (2 bytes)
                 (int)((tensor_d0 & 0xFFFFu) << 16),                 // tensor_dim0 lo
                 (int)((tensor_d0 >> 16) | ((tensor_d1 & 0xFFFFu) << 16)),
                 (int)((tensor_d1 >> 16) | (tile_d0 << 16)),         // tile_dim0
                 (int)(tile_d1),                                     // tile_dim1 (tile_dim2=0)
                 (int)(stride0),                                     // dim0 stride lo
                 0, 0 };
    i32x4 z4 = {0, 0, 0, 0};
#if __clang_major__ >= 23
    i32x8 z8 = {0, 0, 0, 0, 0, 0, 0, 0};
    __builtin_amdgcn_tensor_load_to_lds(g0, g1, z4, z4, z8, 0);
#else
    __builtin_amdgcn_tensor_load_to_lds(g0, g1, z4, z4, 0);
#endif
}

// ---------------- conversion kernels ----------------------------------------

__global__ void k_cvt_bf16(const float* __restrict__ src,
                           unsigned short* __restrict__ dst, int n) {
    int i = blockIdx.x * blockDim.x + threadIdx.x;
    if (i < n) dst[i] = f32_to_bf16(src[i]);
}

// src: [K,N] f32 row-major  ->  dst: [N,K] bf16 row-major
__global__ void k_transpose_bf16(const float* __restrict__ src,
                                 unsigned short* __restrict__ dst,
                                 int K, int N) {
    int i = blockIdx.x * blockDim.x + threadIdx.x;
    if (i < K * N) {
        int nn = i / K, kk = i - nn * K;
        dst[nn * K + kk] = f32_to_bf16(src[kk * N + nn]);
    }
}

// ---------------- QKV projection GEMM ---------------------------------------
// C[8192,768] = Xbf[8192,256] @ Wqkv + bias, scattered into head-major
// Q,K ([bh,2048,32] bf16) and transposed V ([bh,32,2048] bf16).
// Each wave computes a 32x64 block (2x4 WMMA subtiles), 4 waves/block.

__global__ void k_qkv_gemm(const unsigned short* __restrict__ xb,
                           const unsigned short* __restrict__ wT,   // [768,256]
                           const float* __restrict__ bias,          // [768]
                           unsigned short* __restrict__ qbuf,
                           unsigned short* __restrict__ kbuf,
                           unsigned short* __restrict__ vbuf) {
    const int NB = 12;                                // 768/64
    int wave = threadIdx.x >> 5;
    int blk  = blockIdx.x * 4 + wave;                 // 3072 wave-blocks
    int bn = blk % NB, bm = blk / NB;
    int lane = threadIdx.x & 31;
    int half = lane >> 4, l16 = lane & 15;

    FragC acc[2][4] = {};
    for (int k0 = 0; k0 < 256; k0 += 32) {
        FragAB a[2], b[4];
#pragma unroll
        for (int mi = 0; mi < 2; ++mi) {
            const unsigned short* ap =
                xb + (bm * 32 + mi * 16 + l16) * 256 + k0 + half * 8;
            a[mi].q[0] = *(const uint4*)(ap);
            a[mi].q[1] = *(const uint4*)(ap + 16);
        }
#pragma unroll
        for (int nj = 0; nj < 4; ++nj) {
            const unsigned short* bp =
                wT + (bn * 64 + nj * 16 + l16) * 256 + k0 + half * 8;
            b[nj].q[0] = *(const uint4*)(bp);
            b[nj].q[1] = *(const uint4*)(bp + 16);
        }
#pragma unroll
        for (int mi = 0; mi < 2; ++mi)
#pragma unroll
            for (int nj = 0; nj < 4; ++nj)
                acc[mi][nj].v = __builtin_amdgcn_wmma_f32_16x16x32_bf16(
                    false, a[mi].v, false, b[nj].v, (short)0, acc[mi][nj].v,
                    false, false);
    }

#pragma unroll
    for (int nj = 0; nj < 4; ++nj) {
        int col = bn * 64 + nj * 16 + l16;
        float bv   = bias[col];
        int part   = col >> 8;           // 0=Q 1=K 2=V (uniform per subtile)
        int head   = (col & 255) >> 5;   // uniform per subtile
        int within = col & 31;
#pragma unroll
        for (int mi = 0; mi < 2; ++mi)
#pragma unroll
            for (int r = 0; r < 8; ++r) {
                int mrow  = bm * 32 + mi * 16 + half * 8 + r;
                int batch = mrow >> 11, rowin = mrow & 2047;
                int bh    = batch * 8 + head;
                unsigned short hv = f32_to_bf16(acc[mi][nj].f[r] + bv);
                if (part == 0)
                    qbuf[((size_t)bh * 2048 + rowin) * 32 + within] = hv;
                else if (part == 1)
                    kbuf[((size_t)bh * 2048 + rowin) * 32 + within] = hv;
                else
                    vbuf[((size_t)bh * 32 + within) * 2048 + rowin] = hv;
            }
    }
}

// ---------------- flash attention -------------------------------------------
// grid (32,32), 128 threads.  blockIdx.y = bh; the 4 waves handle 4 adjacent
// 16-query tiles of the same head, so K/V tiles are shared: wave 0 TDM-loads
// the 32x32 K tile and 32x32 V^T tile into LDS each iteration.

__global__ void k_attn(const unsigned short* __restrict__ qbuf,  // [bh,2048,32]
                       const unsigned short* __restrict__ kbuf,  // [bh,2048,32]
                       const unsigned short* __restrict__ vbuf,  // [bh,32,2048]
                       unsigned short* __restrict__ obuf) {      // [b,2048,256]
    __shared__ __align__(16) unsigned short ldsK[32 * 32];   // [key][dim]
    __shared__ __align__(16) unsigned short ldsV[32 * 32];   // [dim][key]
    __shared__ __align__(16) unsigned short pLds[4][16 * 32];
    int wave = threadIdx.x >> 5;
    int lane = threadIdx.x & 31;
    int half = lane >> 4, l16 = lane & 15;
    int bh = blockIdx.y;
    int qt = blockIdx.x * 4 + wave;

    const unsigned short* Q  = qbuf + (size_t)bh * 2048 * 32;
    const unsigned short* Kp = kbuf + (size_t)bh * 2048 * 32;
    const unsigned short* Vt = vbuf + (size_t)bh * 32 * 2048;

    FragAB qa;                                        // Q tile: A operand, K=dh=32
    {
        const unsigned short* qp = Q + (qt * 16 + l16) * 32 + half * 8;
        qa.q[0] = *(const uint4*)(qp);
        qa.q[1] = *(const uint4*)(qp + 16);
    }

    float m_i[8], l_i[8];
    FragC o0 = {}, o1 = {};
#pragma unroll
    for (int r = 0; r < 8; ++r) { m_i[r] = -3.0e38f; l_i[r] = 0.f; }
    const float scale = 0.0625f;                      // d^-0.5 = 1/16

    for (int kt = 0; kt < 2048; kt += 32) {
        __syncthreads();   // previous iteration's readers of ldsK/ldsV done
        if (wave == 0) {
            tdm_load_2d((unsigned)(size_t)ldsK, Kp + kt * 32,
                        /*tensor_d0=*/32, /*tensor_d1=*/2048,
                        /*tile_d0=*/32, /*tile_d1=*/32, /*stride0=*/32);
            tdm_load_2d((unsigned)(size_t)ldsV, Vt + kt,
                        /*tensor_d0=*/2048, /*tensor_d1=*/32,
                        /*tile_d0=*/32, /*tile_d1=*/32, /*stride0=*/2048);
            __builtin_amdgcn_s_wait_tensorcnt(0);
        }
        __syncthreads();   // K/V tiles visible to all waves

        // S = Q @ K^T for 32 keys (two 16-key column tiles), K-frags from LDS
        FragAB kb0, kb1;
        {
            const unsigned short* kp0 = &ldsK[l16 * 32 + half * 8];
            kb0.q[0] = *(const uint4*)(kp0);
            kb0.q[1] = *(const uint4*)(kp0 + 16);
            const unsigned short* kp1 = kp0 + 16 * 32;
            kb1.q[0] = *(const uint4*)(kp1);
            kb1.q[1] = *(const uint4*)(kp1 + 16);
        }
        FragC s0 = {}, s1 = {};
        s0.v = __builtin_amdgcn_wmma_f32_16x16x32_bf16(
            false, qa.v, false, kb0.v, (short)0, s0.v, false, false);
        s1.v = __builtin_amdgcn_wmma_f32_16x16x32_bf16(
            false, qa.v, false, kb1.v, (short)0, s1.v, false, false);

        // online softmax over the 32 new columns (rows live in half-waves)
#pragma unroll
        for (int r = 0; r < 8; ++r) {
            float a0 = s0.f[r] * scale, a1 = s1.f[r] * scale;
            float mx = fmaxf(a0, a1);
            mx = fmaxf(mx, __shfl_xor(mx, 1, 32));
            mx = fmaxf(mx, __shfl_xor(mx, 2, 32));
            mx = fmaxf(mx, __shfl_xor(mx, 4, 32));
            mx = fmaxf(mx, __shfl_xor(mx, 8, 32));
            float mn    = fmaxf(m_i[r], mx);
            float alpha = __expf(m_i[r] - mn);
            a0 = __expf(a0 - mn);
            a1 = __expf(a1 - mn);
            float rs = a0 + a1;
            rs += __shfl_xor(rs, 1, 32);
            rs += __shfl_xor(rs, 2, 32);
            rs += __shfl_xor(rs, 4, 32);
            rs += __shfl_xor(rs, 8, 32);
            l_i[r] = l_i[r] * alpha + rs;
            m_i[r] = mn;
            o0.f[r] *= alpha;
            o1.f[r] *= alpha;
            pLds[wave][(half * 8 + r) * 32 + l16]      = f32_to_bf16(a0);
            pLds[wave][(half * 8 + r) * 32 + 16 + l16] = f32_to_bf16(a1);
        }
        __syncthreads();   // P visible (and fences pLds type-punned accesses)

        FragAB pa;
        {
            const unsigned short* pp = &pLds[wave][l16 * 32 + half * 8];
            pa.q[0] = *(const uint4*)(pp);
            pa.q[1] = *(const uint4*)(pp + 16);
        }
        FragAB vb0, vb1;
        {
            const unsigned short* vp0 = &ldsV[l16 * 32 + half * 8];
            vb0.q[0] = *(const uint4*)(vp0);
            vb0.q[1] = *(const uint4*)(vp0 + 16);
            const unsigned short* vp1 = vp0 + 16 * 32;
            vb1.q[0] = *(const uint4*)(vp1);
            vb1.q[1] = *(const uint4*)(vp1 + 16);
        }
        o0.v = __builtin_amdgcn_wmma_f32_16x16x32_bf16(
            false, pa.v, false, vb0.v, (short)0, o0.v, false, false);
        o1.v = __builtin_amdgcn_wmma_f32_16x16x32_bf16(
            false, pa.v, false, vb1.v, (short)0, o1.v, false, false);
    }

    int batch = bh >> 3, head = bh & 7;
#pragma unroll
    for (int r = 0; r < 8; ++r) {
        float inv = 1.0f / l_i[r];
        int qrow = qt * 16 + half * 8 + r;
        size_t base = ((size_t)batch * 2048 + qrow) * 256 + head * 32;
        obuf[base + l16]      = f32_to_bf16(o0.f[r] * inv);
        obuf[base + 16 + l16] = f32_to_bf16(o1.f[r] * inv);
    }
}

// ---------------- output projection GEMM ------------------------------------
// out[8192,256] = A[8192,256] @ Wout + b_out, 32x64 block per wave.

__global__ void k_out_gemm(const unsigned short* __restrict__ ab,   // [8192,256]
                           const unsigned short* __restrict__ wT,   // [256,256]
                           const float* __restrict__ bias,          // [256]
                           float* __restrict__ out) {               // [8192,256]
    const int NB = 4;                                 // 256/64
    int wave = threadIdx.x >> 5;
    int blk  = blockIdx.x * 4 + wave;                 // 1024 wave-blocks
    int bn = blk % NB, bm = blk / NB;
    int lane = threadIdx.x & 31;
    int half = lane >> 4, l16 = lane & 15;

    FragC acc[2][4] = {};
    for (int k0 = 0; k0 < 256; k0 += 32) {
        FragAB a[2], b[4];
#pragma unroll
        for (int mi = 0; mi < 2; ++mi) {
            const unsigned short* ap =
                ab + (bm * 32 + mi * 16 + l16) * 256 + k0 + half * 8;
            a[mi].q[0] = *(const uint4*)(ap);
            a[mi].q[1] = *(const uint4*)(ap + 16);
        }
#pragma unroll
        for (int nj = 0; nj < 4; ++nj) {
            const unsigned short* bp =
                wT + (bn * 64 + nj * 16 + l16) * 256 + k0 + half * 8;
            b[nj].q[0] = *(const uint4*)(bp);
            b[nj].q[1] = *(const uint4*)(bp + 16);
        }
#pragma unroll
        for (int mi = 0; mi < 2; ++mi)
#pragma unroll
            for (int nj = 0; nj < 4; ++nj)
                acc[mi][nj].v = __builtin_amdgcn_wmma_f32_16x16x32_bf16(
                    false, a[mi].v, false, b[nj].v, (short)0, acc[mi][nj].v,
                    false, false);
    }

#pragma unroll
    for (int nj = 0; nj < 4; ++nj) {
        int col = bn * 64 + nj * 16 + l16;
        float bv = bias[col];
#pragma unroll
        for (int mi = 0; mi < 2; ++mi)
#pragma unroll
            for (int r = 0; r < 8; ++r) {
                int mrow = bm * 32 + mi * 16 + half * 8 + r;
                out[(size_t)mrow * 256 + col] = acc[mi][nj].f[r] + bv;
            }
    }
}

// ---------------- host launch ------------------------------------------------

extern "C" void kernel_launch(void* const* d_in, const int* in_sizes, int n_in,
                              void* d_out, int out_size, void* d_ws, size_t ws_size,
                              hipStream_t stream) {
    const float* x     = (const float*)d_in[0];   // [4,2048,256]
    const float* w_qkv = (const float*)d_in[1];   // [256,768]
    const float* b_qkv = (const float*)d_in[2];   // [768]
    const float* w_out = (const float*)d_in[3];   // [256,256]
    const float* b_out = (const float*)d_in[4];   // [256]
    float* out = (float*)d_out;

    char* ws = (char*)d_ws;
    size_t off = 0;
    auto alloc = [&](size_t bytes) -> void* {
        void* p = ws + off;
        off += (bytes + 255) & ~(size_t)255;
        return p;
    };
    unsigned short* xb    = (unsigned short*)alloc((size_t)8192 * 256 * 2);
    unsigned short* wqkvT = (unsigned short*)alloc((size_t)768 * 256 * 2);
    unsigned short* woutT = (unsigned short*)alloc((size_t)256 * 256 * 2);
    unsigned short* qbuf  = (unsigned short*)alloc((size_t)32 * 2048 * 32 * 2);
    unsigned short* kbuf  = (unsigned short*)alloc((size_t)32 * 2048 * 32 * 2);
    unsigned short* vbuf  = (unsigned short*)alloc((size_t)32 * 2048 * 32 * 2);
    unsigned short* obuf  = (unsigned short*)alloc((size_t)8192 * 256 * 2);

    k_cvt_bf16<<<8192, 256, 0, stream>>>(x, xb, 8192 * 256);
    k_transpose_bf16<<<768, 256, 0, stream>>>(w_qkv, wqkvT, 256, 768);
    k_transpose_bf16<<<256, 256, 0, stream>>>(w_out, woutT, 256, 256);

    // 256*12 = 3072 wave-blocks of 32x64, 4 waves/block
    k_qkv_gemm<<<768, 128, 0, stream>>>(xb, wqkvT, b_qkv, qbuf, kbuf, vbuf);
    // 128 query tiles x 32 (b*h), 4 waves/block
    k_attn<<<dim3(32, 32), 128, 0, stream>>>(qbuf, kbuf, vbuf, obuf);
    // 256*4 = 1024 wave-blocks of 32x64, 4 waves/block
    k_out_gemm<<<256, 128, 0, stream>>>(obuf, woutT, b_out, out);
}
